// MultiHeadAttention_7499012899625
// MI455X (gfx1250) — matrix operand order repaired and checked by reading, not compiled, not run on hardware
//
#include <hip/hip_runtime.h>
#include <math.h>

#define B_   2
#define S_   2048
#define D_   512
#define H_   8
#define DK_  64
#define BS_  (B_ * S_)          // 4096
#define INV_TEMP 0.04419417382415922f   // 1/sqrt(512)
#define LN_EPS 1e-5f

#define SC_LD 2052              // padded LDS row stride (floats): 4-bank shift/row
#define XB_LD 516               // padded LDS row stride for fc+LN buffer

typedef __attribute__((ext_vector_type(16))) __bf16 v16bf;
typedef __attribute__((ext_vector_type(8)))  float  v8f;
typedef __attribute__((ext_vector_type(4)))  float  f4;

// ---------------------------------------------------------------------------
// CDNA5 WMMA 16x16x32 bf16 operand layouts (CDNA5 ISA 7.12.2)
// A 16x32: lane L(0-15) row M=L : VGPR0..3 K=0..7, VGPR4..7 K=16..23
//          lanes 16-31 same rows, K offset +8
// B 32x16: lane = col N; this code loads A and B both in k = 16*(lane/16) + s order
// C/D 16x16: VGPR d -> row M = d + 8*(lane/16), col N = lane%16
// ---------------------------------------------------------------------------

__device__ __forceinline__ int a_kbase(int v, int half_sel) {
    return 2 * v + 16 * half_sel;
}

// A from fp32 row-major [ld] (global or LDS), converting to bf16
__device__ __forceinline__ v16bf load_a_f32(const float* A, int ld,
                                            int row, int k0, int half_sel) {
    v16bf a;
    const float* p = A + (size_t)row * ld + k0;
#pragma unroll
    for (int v = 0; v < 8; ++v) {
        int kb = a_kbase(v, half_sel);
        a[2 * v]     = (__bf16)p[kb];
        a[2 * v + 1] = (__bf16)p[kb + 1];
    }
    return a;
}

// A from bf16 row-major [ld]
__device__ __forceinline__ v16bf load_a_bf16(const __bf16* __restrict__ A, int ld,
                                             int row, int k0, int half_sel) {
    v16bf a;
    const __bf16* p = A + (size_t)row * ld + k0;
#pragma unroll
    for (int v = 0; v < 8; ++v) {
        int kb = a_kbase(v, half_sel);
        a[2 * v]     = p[kb];
        a[2 * v + 1] = p[kb + 1];
    }
    return a;
}

// B for "transposed" pattern: Bsrc is NxK row-major, B[k][n] = Bsrc[n][k].
// Per-lane: 16 consecutive values of row n -> vectorizes to wide loads.
__device__ __forceinline__ v16bf load_bt_f32(const float* __restrict__ Bs, int ld,
                                             int ncol, int k0, int half_sel) {
    v16bf b;
    const float* p = Bs + (size_t)ncol * ld + k0 + 16 * half_sel;
#pragma unroll
    for (int v = 0; v < 8; ++v) {
        b[2 * v]     = (__bf16)p[2 * v];
        b[2 * v + 1] = (__bf16)p[2 * v + 1];
    }
    return b;
}

__device__ __forceinline__ v16bf load_bt_bf16(const __bf16* __restrict__ Bs, int ld,
                                              int ncol, int k0, int half_sel) {
    v16bf b;
    const __bf16* p = Bs + (size_t)ncol * ld + k0 + 16 * half_sel;
#pragma unroll
    for (int v = 0; v < 8; ++v) {
        b[2 * v]     = p[2 * v];
        b[2 * v + 1] = p[2 * v + 1];
    }
    return b;
}

#define WMMA_BF16(a, b, c) \
    __builtin_amdgcn_wmma_f32_16x16x32_bf16(false, (a), false, (b), (short)0, (c), false, false)

// ---------------------------------------------------------------------------
// Kernel 1: projection  out = (x @ W.T + bias) * oscale, split heads, bf16.
// oscale = 1/sqrt(D) for Q (folds the score scaling out of the attention
// inner loop), 1.0 for K/V.
// VTRANS==0: outh layout [b, h, s, dk]          (for Q, K)
// VTRANS==1: outh layout [b, h, dk, s] (V^T)    (contiguous B-loads in attn@V)
// One wave = 16x64 strip (4 N-tiles, A fragment hoisted). 2048 waves total.
// ---------------------------------------------------------------------------
template <int VTRANS>
__global__ void qkv_kernel(const float* __restrict__ x, const float* __restrict__ W,
                           const float* __restrict__ bias, __bf16* __restrict__ outh,
                           float oscale) {
    int wave = threadIdx.x >> 5, lane = threadIdx.x & 31;
    int half_sel = lane >> 4, l16 = lane & 15;
    int t = blockIdx.x * 8 + wave;           // 0..2047
    int mt = t >> 3, ng = t & 7;             // 256 m-tiles x 8 n-groups(64)
    int m0 = mt * 16, n0 = ng * 64;

    v8f c0 = {}, c1 = {}, c2 = {}, c3 = {};
    for (int k0 = 0; k0 < D_; k0 += 32) {
        v16bf a  = load_a_f32(x, D_, m0 + l16, k0, half_sel);
        v16bf b0 = load_bt_f32(W, D_, n0 +  0 + l16, k0, half_sel);
        v16bf b1 = load_bt_f32(W, D_, n0 + 16 + l16, k0, half_sel);
        v16bf b2 = load_bt_f32(W, D_, n0 + 32 + l16, k0, half_sel);
        v16bf b3 = load_bt_f32(W, D_, n0 + 48 + l16, k0, half_sel);
        c0 = WMMA_BF16(a, b0, c0);
        c1 = WMMA_BF16(a, b1, c1);
        c2 = WMMA_BF16(a, b2, c2);
        c3 = WMMA_BF16(a, b3, c3);
    }
    int h = ng;                              // n-group == head (64-wide)
    int bidx = m0 >> 11, s0 = (m0 & 2047) + 8 * half_sel;  // tile within one batch
    v8f acc[4] = {c0, c1, c2, c3};
#pragma unroll
    for (int j = 0; j < 4; ++j) {
        int dk = j * 16 + l16;
        float bval = bias[n0 + dk];
        if (VTRANS == 0) {
            __bf16* dst = outh + ((size_t)((bidx * H_ + h) * S_) + s0) * DK_ + dk;
#pragma unroll
            for (int d = 0; d < 8; ++d)
                dst[(size_t)d * DK_] = (__bf16)((acc[j][d] + bval) * oscale);
        } else {
            __bf16* dst = outh + ((size_t)((bidx * H_ + h) * DK_) + dk) * S_ + s0;
#pragma unroll
            for (int d = 0; d < 8; ++d)
                dst[d] = (__bf16)((acc[j][d] + bval) * oscale);   // contiguous b128
        }
    }
}

// ---------------------------------------------------------------------------
// Kernel 2 (fused): for one (h,b) and 16 query rows:
//   phase 1: scores strip 16x2048 = Qs @ K^T -> LDS  (Q pre-scaled by 1/sqrt(D))
//   phase 2: row max/exp/sum in LDS; attn output = exp * rowinv, NT float4 store
//   phase 3: exp(LDS) @ V^T, K split across 8 waves, LDS reduce * rowinv -> ctx
// Dynamic LDS ~161 KB -> 1 workgroup per WGP (320 KB LDS).
// ---------------------------------------------------------------------------
__global__ void fused_attn_kernel(const __bf16* __restrict__ qh,
                                  const __bf16* __restrict__ kh,
                                  const __bf16* __restrict__ vt,
                                  float* __restrict__ attn,
                                  float* __restrict__ ctx) {
    extern __shared__ char smem_raw[];
    float* sc     = (float*)smem_raw;        // [16][SC_LD]
    float* part   = sc + 16 * SC_LD;         // [8][16*64]
    float* red    = part + 8 * 1024;         // [256]
    float* rowmax = red + 256;               // [16]
    float* rowinv = rowmax + 16;             // [16]

    int tid = threadIdx.x;
    int wave = tid >> 5, lane = tid & 31;
    int half_sel = lane >> 4, l16 = lane & 15;
    int hb = blockIdx.y;                     // h*B + b (torch attn layout)
    int h = hb >> 1, b = hb & 1;
    int m0 = blockIdx.x * 16;

    const __bf16* Q = qh + (size_t)((b * H_ + h) * S_) * DK_;
    const __bf16* K = kh + (size_t)((b * H_ + h) * S_) * DK_;
    const __bf16* V = vt + (size_t)((b * H_ + h) * DK_) * S_;   // [64][2048]

    // ---- phase 1: scores into LDS --------------------------------------
    v16bf aq0 = load_a_bf16(Q, DK_, m0 + l16, 0,  half_sel);
    v16bf aq1 = load_a_bf16(Q, DK_, m0 + l16, 32, half_sel);
    for (int ntile = wave; ntile < 128; ntile += 8) {
        int n0 = ntile * 16;
        v16bf bk0 = load_bt_bf16(K, DK_, n0 + l16, 0,  half_sel);
        v16bf bk1 = load_bt_bf16(K, DK_, n0 + l16, 32, half_sel);
        v8f c = {};
        c = WMMA_BF16(aq0, bk0, c);
        c = WMMA_BF16(aq1, bk1, c);
#pragma unroll
        for (int d = 0; d < 8; ++d)
            sc[(d + 8 * half_sel) * SC_LD + n0 + l16] = c[d];
    }
    __syncthreads();

    // ---- phase 2: softmax stats (16 threads per row, float4) -----------
    {
        int grp = tid >> 4, tl = tid & 15;
        float* row = sc + grp * SC_LD;
        float m = -3.4e38f;
        for (int i = tl * 4; i < S_; i += 64) {
            f4 vv = *(const f4*)&row[i];
            m = fmaxf(m, fmaxf(fmaxf(vv.x, vv.y), fmaxf(vv.z, vv.w)));
        }
        red[grp * 16 + tl] = m; __syncthreads();
        for (int s = 8; s > 0; s >>= 1) {
            if (tl < s) red[grp * 16 + tl] = fmaxf(red[grp * 16 + tl], red[grp * 16 + tl + s]);
            __syncthreads();
        }
        if (tl == 0) rowmax[grp] = red[grp * 16];
        __syncthreads();
        float mr = rowmax[grp];
        float sum = 0.f;
        for (int i = tl * 4; i < S_; i += 64) {
            f4 vv = *(const f4*)&row[i];
            vv.x = __expf(vv.x - mr); vv.y = __expf(vv.y - mr);
            vv.z = __expf(vv.z - mr); vv.w = __expf(vv.w - mr);
            *(f4*)&row[i] = vv;                     // keep UNNORMALIZED exp
            sum += vv.x + vv.y + vv.z + vv.w;
        }
        red[grp * 16 + tl] = sum; __syncthreads();
        for (int s = 8; s > 0; s >>= 1) {
            if (tl < s) red[grp * 16 + tl] += red[grp * 16 + tl + s];
            __syncthreads();
        }
        if (tl == 0) rowinv[grp] = 1.0f / red[grp * 16];
        __syncthreads();
    }

    // normalized attn output: float4, non-temporal (268 MB streamed past L2,
    // never re-read from HBM -- phase 3 consumes the LDS copy)
    {
        float* out = attn + (size_t)hb * S_ * S_ + (size_t)m0 * S_;
        for (int idx = tid; idx < 16 * (S_ / 4); idx += 256) {
            int r = idx >> 9, c4 = (idx & 511) * 4;
            f4 vv = *(const f4*)&sc[r * SC_LD + c4];
            float inv = rowinv[r];
            f4 o = vv * inv;
            __builtin_nontemporal_store(o, (f4*)&out[(size_t)r * S_ + c4]);
        }
    }
    __syncthreads();

    // ---- phase 3: exp @ V, K-chunk of 256 per wave ---------------------
    {
        int kbase = wave * 256;
        v8f c0 = {}, c1 = {}, c2 = {}, c3 = {};
        for (int k0 = kbase; k0 < kbase + 256; k0 += 32) {
            v16bf a  = load_a_f32(sc, SC_LD, l16, k0, half_sel);   // LDS read
            v16bf b0 = load_bt_bf16(V, S_,  0 + l16, k0, half_sel);
            v16bf b1 = load_bt_bf16(V, S_, 16 + l16, k0, half_sel);
            v16bf b2 = load_bt_bf16(V, S_, 32 + l16, k0, half_sel);
            v16bf b3 = load_bt_bf16(V, S_, 48 + l16, k0, half_sel);
            c0 = WMMA_BF16(a, b0, c0);
            c1 = WMMA_BF16(a, b1, c1);
            c2 = WMMA_BF16(a, b2, c2);
            c3 = WMMA_BF16(a, b3, c3);
        }
        float* pw = part + wave * 1024;      // [16][64]
#pragma unroll
        for (int d = 0; d < 8; ++d) {
            int r = d + 8 * half_sel;
            pw[r * 64 +  0 + l16] = c0[d];
            pw[r * 64 + 16 + l16] = c1[d];
            pw[r * 64 + 32 + l16] = c2[d];
            pw[r * 64 + 48 + l16] = c3[d];
        }
    }
    __syncthreads();

    // final cross-wave reduce (fold in softmax normalization) -> ctx
    for (int idx = tid; idx < 1024; idx += 256) {
        float s = 0.f;
#pragma unroll
        for (int w = 0; w < 8; ++w) s += part[w * 1024 + idx];
        int r = idx >> 6, n = idx & 63;
        ctx[(size_t)(b * S_ + m0 + r) * D_ + h * DK_ + n] = s * rowinv[r];
    }
}

// ---------------------------------------------------------------------------
// Kernel 3 (fused): x = ctx @ Wfc.T + bfc + residual; y = LayerNorm(x)
// Block = 16 rows x 512 cols: 8 waves x 4 N-tiles (A hoisted), LDS row buffer.
// ---------------------------------------------------------------------------
__global__ void fc_ln_kernel(const float* __restrict__ ctx, const float* __restrict__ W,
                             const float* __restrict__ bias, const float* __restrict__ resid,
                             const float* __restrict__ gamma, const float* __restrict__ beta,
                             float* __restrict__ y) {
    __shared__ float xbuf[16 * XB_LD];
    __shared__ float red[256], red2[256];
    __shared__ float rowmu[16], rowrs[16];

    int tid = threadIdx.x;
    int wave = tid >> 5, lane = tid & 31;
    int half_sel = lane >> 4, l16 = lane & 15;
    int m0 = blockIdx.x * 16;
    int n0 = wave * 64;

    v8f c0 = {}, c1 = {}, c2 = {}, c3 = {};
    for (int k0 = 0; k0 < D_; k0 += 32) {
        v16bf a  = load_a_f32(ctx, D_, m0 + l16, k0, half_sel);
        v16bf b0 = load_bt_f32(W, D_, n0 +  0 + l16, k0, half_sel);
        v16bf b1 = load_bt_f32(W, D_, n0 + 16 + l16, k0, half_sel);
        v16bf b2 = load_bt_f32(W, D_, n0 + 32 + l16, k0, half_sel);
        v16bf b3 = load_bt_f32(W, D_, n0 + 48 + l16, k0, half_sel);
        c0 = WMMA_BF16(a, b0, c0);
        c1 = WMMA_BF16(a, b1, c1);
        c2 = WMMA_BF16(a, b2, c2);
        c3 = WMMA_BF16(a, b3, c3);
    }
    v8f acc[4] = {c0, c1, c2, c3};
#pragma unroll
    for (int j = 0; j < 4; ++j) {
        int n = n0 + j * 16 + l16;
        float bval = bias[n];
#pragma unroll
        for (int d = 0; d < 8; ++d) {
            int r = d + 8 * half_sel;
            xbuf[r * XB_LD + n] = acc[j][d] + bval + resid[(size_t)(m0 + r) * D_ + n];
        }
    }
    __syncthreads();

    // mean / var per row (16 threads per row; E[x^2]-mu^2)
    {
        int grp = tid >> 4, tl = tid & 15;
        const float* row = xbuf + grp * XB_LD;
        float s = 0.f, s2 = 0.f;
        for (int i = tl * 4; i < D_; i += 64) {
            f4 xv = *(const f4*)&row[i];
            s  += xv.x + xv.y + xv.z + xv.w;
            s2 += xv.x * xv.x + xv.y * xv.y + xv.z * xv.z + xv.w * xv.w;
        }
        red[grp * 16 + tl] = s; red2[grp * 16 + tl] = s2; __syncthreads();
        for (int st = 8; st > 0; st >>= 1) {
            if (tl < st) {
                red[grp * 16 + tl]  += red[grp * 16 + tl + st];
                red2[grp * 16 + tl] += red2[grp * 16 + tl + st];
            }
            __syncthreads();
        }
        if (tl == 0) {
            float mu  = red[grp * 16] * (1.0f / D_);
            float var = red2[grp * 16] * (1.0f / D_) - mu * mu;
            rowmu[grp] = mu;
            rowrs[grp] = rsqrtf(var + LN_EPS);
        }
        __syncthreads();
    }

    // write y (float4)
    for (int idx = tid; idx < 16 * (D_ / 4); idx += 256) {
        int r = idx >> 7, c4 = (idx & 127) * 4;
        f4 xv = *(const f4*)&xbuf[r * XB_LD + c4];
        float mu = rowmu[r], rs = rowrs[r];
        f4 g  = *(const f4*)&gamma[c4];
        f4 bt = *(const f4*)&beta[c4];
        f4 o = (xv - mu) * rs * g + bt;
        *(f4*)&y[(size_t)(m0 + r) * D_ + c4] = o;
    }
}

// ---------------------------------------------------------------------------
extern "C" void kernel_launch(void* const* d_in, const int* in_sizes, int n_in,
                              void* d_out, int out_size, void* d_ws, size_t ws_size,
                              hipStream_t stream) {
    const float* q     = (const float*)d_in[0];
    const float* k     = (const float*)d_in[1];
    const float* v     = (const float*)d_in[2];
    const float* Wq    = (const float*)d_in[3];
    const float* bq    = (const float*)d_in[4];
    const float* Wk    = (const float*)d_in[5];
    const float* bk    = (const float*)d_in[6];
    const float* Wv    = (const float*)d_in[7];
    const float* bv    = (const float*)d_in[8];
    const float* Wfc   = (const float*)d_in[9];
    const float* bfc   = (const float*)d_in[10];
    const float* gamma = (const float*)d_in[11];
    const float* beta  = (const float*)d_in[12];

    float* y    = (float*)d_out;
    float* attn = y + (size_t)BS_ * D_;          // 67,108,864 floats

    char* ws = (char*)d_ws;
    __bf16* qh  = (__bf16*)(ws);                 //  4 MB   [b,h,s,dk] (pre-scaled)
    __bf16* kh  = (__bf16*)(ws + (4u  << 20));   //  4 MB   [b,h,s,dk]
    __bf16* vt  = (__bf16*)(ws + (8u  << 20));   //  4 MB   [b,h,dk,s]  (V^T)
    float*  ctx = (float*) (ws + (12u << 20));   //  8 MB  (20 MB total)

    const size_t SMEM = (size_t)(16 * SC_LD + 8 * 1024 + 256 + 32) * sizeof(float);

    qkv_kernel<0><<<dim3(256), 256, 0, stream>>>(q, Wq, bq, qh, INV_TEMP);
    qkv_kernel<0><<<dim3(256), 256, 0, stream>>>(k, Wk, bk, kh, 1.0f);
    qkv_kernel<1><<<dim3(256), 256, 0, stream>>>(v, Wv, bv, vt, 1.0f);
    fused_attn_kernel<<<dim3(128, 16), 256, SMEM, stream>>>(qh, kh, vt, attn, ctx);
    fc_ln_kernel<<<dim3(256), 256, 0, stream>>>(ctx, Wfc, bfc, q, gamma, beta, y);
}
